// AttentionModule_36215164240033
// MI455X (gfx1250) — matrix operand order repaired and checked by reading, not compiled, not run on hardware
//
#include <hip/hip_runtime.h>
#include <hip/hip_bf16.h>

typedef __attribute__((ext_vector_type(16))) _Float16 v16h;
typedef __attribute__((ext_vector_type(8)))  _Float16 v8h;
typedef __attribute__((ext_vector_type(4)))  _Float16 v4h;
typedef __attribute__((ext_vector_type(8)))  float    v8f;
typedef __attribute__((ext_vector_type(4)))  float    v4f;

// Problem constants
#define BSZ 8
#define SEQ 1024
#define DM  512
#define NH  8
#define DKH 64
#define MTOT (BSZ*SEQ)          // 8192

// Workspace byte offsets
#define OFF_XPE  (size_t)0            // 8192*512 f16  = 8 MB
#define OFF_Q    ((size_t)8<<20)      // 8 MB
#define OFF_KT   ((size_t)16<<20)     // 8 MB
#define OFF_V    ((size_t)24<<20)     // 8 MB
#define OFF_AO   ((size_t)32<<20)     // 8 MB
#define OFF_YPRE ((size_t)8<<20)      // 16 MB (overlays Q/KT, both dead by then)
#define OFF_SB   ((size_t)40<<20)     // 8192 f32

static __device__ __forceinline__ v16h cat8(v8h lo, v8h hi) {
  return __builtin_shufflevector(lo, hi, 0,1,2,3,4,5,6,7,8,9,10,11,12,13,14,15);
}

// ---------------- Kernel 1: pos-encoding + boundary predictor ----------------
__global__ void k_prep(const float* __restrict__ x,
                       const float* __restrict__ w1, const float* __restrict__ b1,
                       const float* __restrict__ w2, const float* __restrict__ b2,
                       _Float16* __restrict__ xpe16, float* __restrict__ sb)
{
  __shared__ float xrow[DM];
  __shared__ float red[256];
  const int row = blockIdx.x;                 // 0..8191
  const int t   = threadIdx.x;                // 0..255
  const int s   = row & (SEQ - 1);
  const float kdiv = -9.210340371976184f / (float)DM;   // -ln(10000)/D
  for (int c = t; c < DM; c += 256) {
    float ang = (float)s * __expf((float)(c & ~1) * kdiv);
    float pe  = (c & 1) ? __cosf(ang) : __sinf(ang);
    float v   = x[(size_t)row * DM + c] + pe;
    xrow[c] = v;
    xpe16[(size_t)row * DM + c] = (_Float16)v;
  }
  __syncthreads();
  // h[t] = relu(xrow . w1[:,t] + b1[t]);  partial = h[t] * w2[t]
  float acc = b1[t];
  for (int k = 0; k < DM; ++k) acc = fmaf(xrow[k], w1[k * 256 + t], acc);
  float h = fmaxf(acc, 0.0f);
  red[t] = h * w2[t];
  __syncthreads();
  for (int off = 128; off > 0; off >>= 1) {
    if (t < off) red[t] += red[t + off];
    __syncthreads();
  }
  if (t == 0) {
    float z = (red[0] + b2[0]) * 0.5f;        // / TEMP(=2)
    sb[row] = 1.0f / (1.0f + __expf(-z));
  }
}

// ---------------- Kernel 2: QKV projection GEMM (WMMA, LDS staged) ----------------
__global__ void k_qkv(const _Float16* __restrict__ xpe,
                      const float* __restrict__ wq, const float* __restrict__ bq,
                      const float* __restrict__ wk, const float* __restrict__ bk,
                      const float* __restrict__ wv, const float* __restrict__ bv,
                      _Float16* __restrict__ Q16, _Float16* __restrict__ KT16,
                      _Float16* __restrict__ V16)
{
  const int zz = blockIdx.z;
  const float* W    = (zz == 0) ? wq : (zz == 1) ? wk : wv;
  const float* bias = (zz == 0) ? bq : (zz == 1) ? bk : bv;
  __shared__ __align__(32) _Float16 As[128 * 32];
  __shared__ __align__(32) _Float16 Bs[32 * 64];
  const int t = threadIdx.x, lane = t & 31, wave = t >> 5;
  const int mb = blockIdx.x * 128, nb = blockIdx.y * 64;
  v8f acc[4] = {};
  const int m = lane & 15, kh = lane >> 4;
  for (int kb = 0; kb < DM; kb += 32) {
    // A tile: 128x32 f16 — two statically-indexed chunks so both loads issue first
    {
      int l0 = t * 8,         m0 = l0 >> 5, k0 = l0 & 31;
      int l1 = (t + 256) * 8, m1 = l1 >> 5, k1 = l1 & 31;
      v8h a0 = *(const v8h*)&xpe[(size_t)(mb + m0) * DM + kb + k0];
      v8h a1 = *(const v8h*)&xpe[(size_t)(mb + m1) * DM + kb + k1];
      *(v8h*)&As[m0 * 32 + k0] = a0;
      *(v8h*)&As[m1 * 32 + k1] = a1;
    }
    {                                             // B tile: 32x64, f32->f16
      int lin = t * 8, kk = lin >> 6, nn = lin & 63;
      const float* src = &W[(size_t)(kb + kk) * DM + nb + nn];
      v8h tmp;
      #pragma unroll
      for (int i = 0; i < 8; ++i) tmp[i] = (_Float16)src[i];
      *(v8h*)&Bs[kk * 64 + nn] = tmp;
    }
    __syncthreads();
    const _Float16* arow = &As[(wave * 16 + m) * 32 + kh * 8];
    v16h afrag = cat8(*(const v8h*)arow, *(const v8h*)(arow + 16));
    #pragma unroll
    for (int nt = 0; nt < 4; ++nt) {
      v16h bfrag;
      #pragma unroll
      for (int i = 0; i < 16; ++i) bfrag[i] = Bs[(kh * 8 + (i & 7) + (i >> 3) * 16) * 64 + nt * 16 + m];
      acc[nt] = __builtin_amdgcn_wmma_f32_16x16x32_f16(false, afrag, false, bfrag,
                                                       (short)0, acc[nt], false, false);
    }
    __syncthreads();
  }
  const int m0 = mb + wave * 16 + ((lane >> 4) << 3);
  #pragma unroll
  for (int nt = 0; nt < 4; ++nt) {
    int ng = nb + nt * 16 + (lane & 15);
    float bv_ = bias[ng];
    int hh = ng >> 6, dk = ng & 63;
    #pragma unroll
    for (int r = 0; r < 8; ++r) {
      int mg = m0 + r;
      int bi = mg >> 10, si = mg & 1023;
      _Float16 val = (_Float16)(acc[nt][r] + bv_);
      if (zz == 0)      Q16 [((size_t)(bi * NH + hh) * SEQ + si) * DKH + dk] = val;
      else if (zz == 1) KT16[((size_t)(bi * NH + hh) * DKH + dk) * SEQ + si] = val;
      else              V16 [((size_t)(bi * NH + hh) * SEQ + si) * DKH + dk] = val;
    }
  }
}

// ---- Kernel 3 (fused): scores + bias + mask -> softmax -> attn out + attn@V ----
// Block = 8 waves, owns 16 query rows of one (b,h). LDS: f32 score tile (64KB)
// + f16 prob tile (32KB). attn is written to HBM exactly once, non-temporally
// so the 268MB stream does not evict the L2-resident K/V/Q (24MB total).
__global__ void k_attn(const _Float16* __restrict__ Q16, const _Float16* __restrict__ KT16,
                       const _Float16* __restrict__ V16,
                       const float* __restrict__ sb, const unsigned char* __restrict__ mask,
                       float* __restrict__ attn, _Float16* __restrict__ AO16)
{
  __shared__ __align__(32) float    Sf[16 * 1024];   // 64 KB scores (later: AV partials)
  __shared__ __align__(32) _Float16 Sh[16 * 1024];   // 32 KB probs f16
  const int mb = blockIdx.x * 16;
  const int bh = blockIdx.y, bi = bh >> 3, hh = bh & 7;
  const int t = threadIdx.x, lane = t & 31, wave = t >> 5;
  const _Float16* Qb = Q16  + (size_t)bh * SEQ * DKH;
  const _Float16* Kb = KT16 + (size_t)bh * DKH * SEQ;
  const _Float16* Vb = V16  + (size_t)bh * SEQ * DKH;
  float* ab = attn + (size_t)bh * SEQ * SEQ;
  const int m = lane & 15, kh = lane >> 4, r0 = (lane >> 4) << 3;

  // ---- phase 1: scores strip [16 x 128] per wave, via WMMA ----
  {
    v8f acc[8] = {};
    const _Float16* qrow = Qb + (size_t)(mb + m) * DKH;
    #pragma unroll
    for (int ks = 0; ks < DKH; ks += 32) {
      v16h afrag = cat8(*(const v8h*)(qrow + ks + kh * 8),
                        *(const v8h*)(qrow + ks + kh * 8 + 16));
      const _Float16* krow = Kb + (size_t)(ks + lane) * SEQ + wave * 128;
      __builtin_prefetch(krow + 32 * SEQ, 0, 2);   // next k-strip (L2-resident)
      #pragma unroll
      for (int nt = 0; nt < 8; ++nt) {
        v16h bfrag;
        #pragma unroll
        for (int i = 0; i < 16; ++i) bfrag[i] = Kb[(size_t)(ks + kh * 8 + (i & 7) + (i >> 3) * 16) * SEQ + wave * 128 + nt * 16 + m];
        acc[nt] = __builtin_amdgcn_wmma_f32_16x16x32_f16(false, afrag, false, bfrag,
                                                         (short)0, acc[nt], false, false);
      }
    }
    const float* srow = sb + bi * SEQ;
    const unsigned char* mrow = mask + bi * SEQ;
    float sq[8]; bool mq[8];
    #pragma unroll
    for (int r = 0; r < 8; ++r) { sq[r] = srow[mb + r0 + r]; mq[r] = mrow[mb + r0 + r] != 0; }
    #pragma unroll
    for (int nt = 0; nt < 8; ++nt) {
      int ng = wave * 128 + nt * 16 + (lane & 15);
      float sk = srow[ng];
      bool  mk = mrow[ng] != 0;
      #pragma unroll
      for (int r = 0; r < 8; ++r) {
        float v = acc[nt][r] * 0.125f + 0.5f * (sq[r] + sk);
        if (!(mk && mq[r])) v = -1e9f;
        Sf[(r0 + r) * 1024 + ng] = v;
      }
    }
  }
  __syncthreads();

  // ---- phase 2: softmax (wave handles 2 rows); write attn (HBM, once, NT) + f16 probs ----
  #pragma unroll
  for (int rr = 0; rr < 2; ++rr) {
    const int row = wave * 2 + rr;
    const float* p = &Sf[row * 1024];
    v4f v[8];
    float mx = -3.0e38f;
    #pragma unroll
    for (int i = 0; i < 8; ++i) {
      v[i] = *(const v4f*)(p + (i * 32 + lane) * 4);
      mx = fmaxf(mx, fmaxf(fmaxf(v[i].x, v[i].y), fmaxf(v[i].z, v[i].w)));
    }
    #pragma unroll
    for (int off = 16; off > 0; off >>= 1) mx = fmaxf(mx, __shfl_xor(mx, off, 32));
    float sum = 0.0f;
    #pragma unroll
    for (int i = 0; i < 8; ++i) {
      v[i].x = __expf(v[i].x - mx); v[i].y = __expf(v[i].y - mx);
      v[i].z = __expf(v[i].z - mx); v[i].w = __expf(v[i].w - mx);
      sum += v[i].x + v[i].y + v[i].z + v[i].w;
    }
    #pragma unroll
    for (int off = 16; off > 0; off >>= 1) sum += __shfl_xor(sum, off, 32);
    const float inv = 1.0f / sum;
    float* gout = ab + (size_t)(mb + row) * SEQ;
    #pragma unroll
    for (int i = 0; i < 8; ++i) {
      v[i] *= inv;
      __builtin_nontemporal_store(v[i], (v4f*)(gout + (i * 32 + lane) * 4));
      v4h hv; hv[0] = (_Float16)v[i].x; hv[1] = (_Float16)v[i].y;
              hv[2] = (_Float16)v[i].z; hv[3] = (_Float16)v[i].w;
      *(v4h*)&Sh[row * 1024 + (i * 32 + lane) * 4] = hv;
    }
  }
  __syncthreads();

  // ---- phase 3: out[16x64] = probs @ V, K-split across waves ----
  {
    v8f acc[4] = {};
    #pragma unroll
    for (int kk = 0; kk < 4; ++kk) {
      int ks = wave * 128 + kk * 32;
      const _Float16* arow = &Sh[m * 1024 + ks + kh * 8];
      v16h afrag = cat8(*(const v8h*)arow, *(const v8h*)(arow + 16));
      const _Float16* vrow = Vb + (size_t)(ks + lane) * DKH;
      __builtin_prefetch(vrow + 32 * DKH, 0, 2);
      #pragma unroll
      for (int nt = 0; nt < 4; ++nt) {
        v16h bfrag;
        #pragma unroll
        for (int i = 0; i < 16; ++i) bfrag[i] = Vb[(size_t)(ks + kh * 8 + (i & 7) + (i >> 3) * 16) * DKH + nt * 16 + m];
        acc[nt] = __builtin_amdgcn_wmma_f32_16x16x32_f16(false, afrag, false, bfrag,
                                                         (short)0, acc[nt], false, false);
      }
    }
    // stash per-wave partials in (now dead) Sf region: Pp[wave][16][64]
    float* Pp = Sf;
    #pragma unroll
    for (int nt = 0; nt < 4; ++nt)
      #pragma unroll
      for (int r = 0; r < 8; ++r)
        Pp[wave * 1024 + (r0 + r) * 64 + nt * 16 + (lane & 15)] = acc[nt][r];
  }
  __syncthreads();

  // ---- phase 4: cross-wave reduction of AV partials -> AO16 ----
  const float* Pp = Sf;
  #pragma unroll
  for (int i = 0; i < 4; ++i) {
    int idx = t * 4 + i;                 // 0..1023 = mr*64 + dk
    float s = 0.0f;
    #pragma unroll
    for (int w = 0; w < 8; ++w) s += Pp[w * 1024 + idx];
    int mr = idx >> 6, dk = idx & 63;
    AO16[((size_t)bi * SEQ + mb + mr) * DM + hh * DKH + dk] = (_Float16)s;
  }
}

// ---------------- Kernel 4: output projection (WMMA, LDS staged, f32 out) ----------------
__global__ void k_oproj(const _Float16* __restrict__ A, const float* __restrict__ W,
                        const float* __restrict__ bias, float* __restrict__ Y)
{
  __shared__ __align__(32) _Float16 As[128 * 32];
  __shared__ __align__(32) _Float16 Bs[32 * 64];
  const int t = threadIdx.x, lane = t & 31, wave = t >> 5;
  const int mb = blockIdx.x * 128, nb = blockIdx.y * 64;
  v8f acc[4] = {};
  const int m = lane & 15, kh = lane >> 4;
  for (int kb = 0; kb < DM; kb += 32) {
    {
      int l0 = t * 8,         m0 = l0 >> 5, k0 = l0 & 31;
      int l1 = (t + 256) * 8, m1 = l1 >> 5, k1 = l1 & 31;
      v8h a0 = *(const v8h*)&A[(size_t)(mb + m0) * DM + kb + k0];
      v8h a1 = *(const v8h*)&A[(size_t)(mb + m1) * DM + kb + k1];
      *(v8h*)&As[m0 * 32 + k0] = a0;
      *(v8h*)&As[m1 * 32 + k1] = a1;
    }
    {
      int lin = t * 8, kk = lin >> 6, nn = lin & 63;
      const float* src = &W[(size_t)(kb + kk) * DM + nb + nn];
      v8h tmp;
      #pragma unroll
      for (int i = 0; i < 8; ++i) tmp[i] = (_Float16)src[i];
      *(v8h*)&Bs[kk * 64 + nn] = tmp;
    }
    __syncthreads();
    const _Float16* arow = &As[(wave * 16 + m) * 32 + kh * 8];
    v16h afrag = cat8(*(const v8h*)arow, *(const v8h*)(arow + 16));
    #pragma unroll
    for (int nt = 0; nt < 4; ++nt) {
      v16h bfrag;
      #pragma unroll
      for (int i = 0; i < 16; ++i) bfrag[i] = Bs[(kh * 8 + (i & 7) + (i >> 3) * 16) * 64 + nt * 16 + m];
      acc[nt] = __builtin_amdgcn_wmma_f32_16x16x32_f16(false, afrag, false, bfrag,
                                                       (short)0, acc[nt], false, false);
    }
    __syncthreads();
  }
  const int m0 = mb + wave * 16 + ((lane >> 4) << 3);
  #pragma unroll
  for (int nt = 0; nt < 4; ++nt) {
    int ng = nb + nt * 16 + (lane & 15);
    float bv_ = bias[ng];
    #pragma unroll
    for (int r = 0; r < 8; ++r)
      Y[(size_t)(m0 + r) * DM + ng] = acc[nt][r] + bv_;
  }
}

// ---------------- Kernel 5: residual + layernorm (one wave per row) ----------------
__global__ void k_ln(const float* __restrict__ x, const float* __restrict__ ypre,
                     const float* __restrict__ g, const float* __restrict__ b,
                     float* __restrict__ y)
{
  const int t = threadIdx.x, lane = t & 31, wave = t >> 5;
  const int row = blockIdx.x * 8 + wave;                  // 8192 rows
  const float* xr = x    + (size_t)row * DM;
  const float* pr = ypre + (size_t)row * DM;
  float vals[16];
  float sum = 0.0f, sq = 0.0f;
  #pragma unroll
  for (int i = 0; i < 16; ++i) {
    int c = i * 32 + lane;
    float v = xr[c] + pr[c];
    vals[i] = v; sum += v; sq += v * v;
  }
  #pragma unroll
  for (int off = 16; off > 0; off >>= 1) {
    sum += __shfl_xor(sum, off, 32);
    sq  += __shfl_xor(sq,  off, 32);
  }
  const float mu  = sum * (1.0f / DM);
  const float var = sq * (1.0f / DM) - mu * mu;
  const float rs  = rsqrtf(var + 1e-5f);
  #pragma unroll
  for (int i = 0; i < 16; ++i) {
    int c = i * 32 + lane;
    float o = (vals[i] - mu) * rs * g[c] + b[c];
    __builtin_nontemporal_store(o, &y[(size_t)row * DM + c]);
  }
}

extern "C" void kernel_launch(void* const* d_in, const int* in_sizes, int n_in,
                              void* d_out, int out_size, void* d_ws, size_t ws_size,
                              hipStream_t stream) {
  const float*         x    = (const float*)d_in[0];
  const unsigned char* mask = (const unsigned char*)d_in[1];
  const float* wq = (const float*)d_in[2],  *bq = (const float*)d_in[3];
  const float* wk = (const float*)d_in[4],  *bk = (const float*)d_in[5];
  const float* wv = (const float*)d_in[6],  *bv = (const float*)d_in[7];
  const float* wo = (const float*)d_in[8],  *bo = (const float*)d_in[9];
  const float* w1 = (const float*)d_in[10], *b1 = (const float*)d_in[11];
  const float* w2 = (const float*)d_in[12], *b2 = (const float*)d_in[13];
  const float* lng = (const float*)d_in[14], *lnb = (const float*)d_in[15];

  float* y    = (float*)d_out;
  float* attn = y + (size_t)BSZ * SEQ * DM;               // +4,194,304

  char* ws = (char*)d_ws;
  _Float16* XPE  = (_Float16*)(ws + OFF_XPE);
  _Float16* Q16  = (_Float16*)(ws + OFF_Q);
  _Float16* KT16 = (_Float16*)(ws + OFF_KT);
  _Float16* V16  = (_Float16*)(ws + OFF_V);
  _Float16* AO16 = (_Float16*)(ws + OFF_AO);
  float*    YPRE = (float*)(ws + OFF_YPRE);
  float*    SB   = (float*)(ws + OFF_SB);

  k_prep <<<MTOT, 256, 0, stream>>>(x, w1, b1, w2, b2, XPE, SB);
  k_qkv  <<<dim3(MTOT/128, DM/64, 3), 256, 0, stream>>>(XPE, wq, bq, wk, bk, wv, bv, Q16, KT16, V16);
  k_attn <<<dim3(SEQ/16, BSZ*NH), 256, 0, stream>>>(Q16, KT16, V16, SB, mask, attn, AO16);
  k_oproj<<<dim3(MTOT/128, DM/64), 256, 0, stream>>>(AO16, wo, bo, YPRE);
  k_ln   <<<MTOT/8, 256, 0, stream>>>(x, YPRE, lng, lnb, y);
}
